// DictMoEDirect_9861244911919
// MI455X (gfx1250) — compile-verified, matrix operand order checked by reading, not run on hardware
//
#include <hip/hip_runtime.h>
#include <hip/hip_bf16.h>

// ---------------------------------------------------------------------------
// DictMoE "merged expert" forward for MI455X (gfx1250, wave32, WMMA).
//   H=1024, DFF=4096, E=8, B=8, S=512.
// Pipeline:
//   1) x (fp32 [S,B,H]) -> bf16 (row r = s*B+b, so gate GEMM is contiguous)
//   2) gW1 -> bf16
//   3) hgate = relu(x @ gW1^T + gb1)          (WMMA GEMM, M=4096,N=1024,K=1024)
//   4) g[b,e] = mean_s(hgate_row . gW2[e]) + gb2[e]
//   5) W1m/W2m (bf16) = sum_e g[b,e]*eW{1,2}[e]; b1m/b2m likewise (fp32)
//   6) per b: y1 = relu(x_b @ W1m_b^T + b1m_b)   (WMMA, 512x4096, K=1024)
//   7) per b: out = y1 @ W2m_b^T + b2m_b (fp32)  (WMMA, 512x1024, K=4096)
// GEMM: 128x256 block tile, 64x64 wave tile (16 WMMA / 16 ds_load per K-step),
// double-buffered LDS staged with CDNA5 async global->LDS (ASYNCcnt).
// ---------------------------------------------------------------------------

#define Hdim 1024
#define DFF  4096
#define NE   8
#define NB   8
#define NS   512

typedef __attribute__((ext_vector_type(16))) __bf16 v16bf;
typedef __attribute__((ext_vector_type(8)))  float  v8f;

#define TILE_M 128
#define TILE_N 256
#define TILE_K 32
#define LDSS   40      // padded LDS row stride in halves (80 B, keeps 16B align)

union FragU { uint4 q[2]; v16bf v; };

// Async global->LDS, 16 bytes per lane (GLOBAL_LOAD_ASYNC_TO_LDS_B128).
// vdst = 32-bit LDS byte address, vaddr = 64-bit global address (GV mode).
__device__ __forceinline__ void async_copy16(unsigned lds_addr, const void* gptr)
{
    asm volatile("global_load_async_to_lds_b128 %0, %1, off"
                 :: "v"(lds_addr), "v"((unsigned long long)(size_t)gptr)
                 : "memory");
}
// Generic pointer to LDS: low 32 bits are the LDS byte offset (aperture rule).
__device__ __forceinline__ unsigned lds_addr_of(const void* p)
{
    return (unsigned)(size_t)p;
}

// ---------------------------------------------------------------------------
// Tiled GEMM: C[M,N] = A[M,K] * W[N,K]^T + bias[N]; optional ReLU+bf16 store.
// 256 threads = 8 waves; wave grid 2(M) x 4(N); each wave does 4x4 WMMA tiles.
// ---------------------------------------------------------------------------
__global__ __launch_bounds__(256)
void gemm_bf16_wmma(const __bf16* __restrict__ A, int lda,
                    const __bf16* __restrict__ W, int ldw,
                    const float* __restrict__ bias,
                    void* __restrict__ C, int ldc,
                    int K, int relu_bf16_out)
{
    __shared__ __bf16 As[2][TILE_M * LDSS];   // 2 x 10240 B
    __shared__ __bf16 Bs[2][TILE_N * LDSS];   // 2 x 20480 B

    const int tid  = threadIdx.x;
    const int lane = tid & 31;
    const int wid  = tid >> 5;       // 0..7
    const int wm   = wid >> 2;       // wave M index (2 waves, 64 rows each)
    const int wn   = wid & 3;        // wave N index (4 waves, 64 cols each)
    const int half = lane >> 4;      // hi/lo half of the wave
    const int l16  = lane & 15;

    const int blockM = blockIdx.y * TILE_M;
    const int blockN = blockIdx.x * TILE_N;

    // ---- staging geometry: A = 512 chunks (2/thread), B = 1024 chunks (4/thread)
    int arow[2], acol[2];
    #pragma unroll
    for (int i = 0; i < 2; ++i) {
        int lin = tid + i * 256;            // 0..511
        arow[i] = lin >> 2;                 // 0..127
        acol[i] = (lin & 3) << 3;           // halves: 0,8,16,24
    }
    int brow[4], bcol[4];
    #pragma unroll
    for (int i = 0; i < 4; ++i) {
        int lin = tid + i * 256;            // 0..1023
        brow[i] = lin >> 2;                 // 0..255
        bcol[i] = (lin & 3) << 3;
    }

    const __bf16* gA[2];
    const __bf16* gB[4];
    #pragma unroll
    for (int i = 0; i < 2; ++i) gA[i] = A + (size_t)(blockM + arow[i]) * lda + acol[i];
    #pragma unroll
    for (int i = 0; i < 4; ++i) gB[i] = W + (size_t)(blockN + brow[i]) * ldw + bcol[i];

    unsigned lA[2][2], lB[2][4];
    #pragma unroll
    for (int p = 0; p < 2; ++p) {
        #pragma unroll
        for (int i = 0; i < 2; ++i) lA[p][i] = lds_addr_of(&As[p][arow[i] * LDSS + acol[i]]);
        #pragma unroll
        for (int i = 0; i < 4; ++i) lB[p][i] = lds_addr_of(&Bs[p][brow[i] * LDSS + bcol[i]]);
    }

    // ---- prologue: stage K-tile 0 into buffer 0 (6 async ops / thread) ----
    #pragma unroll
    for (int i = 0; i < 2; ++i) async_copy16(lA[0][i], gA[i]);
    #pragma unroll
    for (int i = 0; i < 4; ++i) async_copy16(lB[0][i], gB[i]);

    v8f acc[4][4] = {};
    int cur = 0;

    for (int kt = 0; kt < K; kt += TILE_K) {
        const int ktn  = kt + TILE_K;
        const bool more = ktn < K;

        if (more) {  // stage next tile into the other buffer
            const int nxt = cur ^ 1;
            #pragma unroll
            for (int i = 0; i < 2; ++i) async_copy16(lA[nxt][i], gA[i] + ktn);
            #pragma unroll
            for (int i = 0; i < 4; ++i) async_copy16(lB[nxt][i], gB[i] + ktn);
            // 6 newest may stay in flight; current tile's 6 must be done
            asm volatile("s_wait_asynccnt 0x6" ::: "memory");
        } else {
            asm volatile("s_wait_asynccnt 0x0" ::: "memory");
        }
        __syncthreads();

        // ---- fragments per 16-bit A/B VGPR layout:
        //      lane holds row m (or col n) = lane%16; K = {half*8..+7, 16+half*8..+7}
        const __bf16* Ac = As[cur];
        const __bf16* Bc = Bs[cur];
        v16bf afrag[4];
        #pragma unroll
        for (int mi = 0; mi < 4; ++mi) {
            int m = wm * 64 + mi * 16 + l16;
            FragU u;
            u.q[0] = *(const uint4*)&Ac[m * LDSS + half * 8];
            u.q[1] = *(const uint4*)&Ac[m * LDSS + 16 + half * 8];
            afrag[mi] = u.v;
        }
        v16bf bfrag[4];
        #pragma unroll
        for (int ni = 0; ni < 4; ++ni) {
            int n = wn * 64 + ni * 16 + l16;
            FragU u;
            u.q[0] = *(const uint4*)&Bc[n * LDSS + half * 8];
            u.q[1] = *(const uint4*)&Bc[n * LDSS + 16 + half * 8];
            bfrag[ni] = u.v;
        }

        #pragma unroll
        for (int mi = 0; mi < 4; ++mi)
            #pragma unroll
            for (int ni = 0; ni < 4; ++ni)
                acc[mi][ni] = __builtin_amdgcn_wmma_f32_16x16x32_bf16(
                    false, afrag[mi], false, bfrag[ni],
                    (short)0, acc[mi][ni], false, false);

        __syncthreads();   // reads done before buffer `cur` is re-staged
        cur ^= 1;
    }

    // ---- epilogue: bias (+ReLU) and store. C layout: m=(lane/16)*8+i, n=lane%16
    #pragma unroll
    for (int mi = 0; mi < 4; ++mi) {
        #pragma unroll
        for (int ni = 0; ni < 4; ++ni) {
            int n  = blockN + wn * 64 + ni * 16 + l16;
            float bv = bias[n];
            #pragma unroll
            for (int i = 0; i < 8; ++i) {
                int m = blockM + wm * 64 + mi * 16 + half * 8 + i;
                float v = acc[mi][ni][i] + bv;
                if (relu_bf16_out) {
                    v = v > 0.0f ? v : 0.0f;
                    ((__bf16*)C)[(size_t)m * ldc + n] = (__bf16)v;
                } else {
                    ((float*)C)[(size_t)m * ldc + n] = v;
                }
            }
        }
    }
}

// ---------------------------------------------------------------------------
__global__ __launch_bounds__(256)
void f32_to_bf16_vec4(const float* __restrict__ in, __bf16* __restrict__ out, int n4)
{
    int i = blockIdx.x * 256 + threadIdx.x;
    if (i >= n4) return;
    float4 v = ((const float4*)in)[i];
    alignas(8) __bf16 o[4] = {(__bf16)v.x, (__bf16)v.y, (__bf16)v.z, (__bf16)v.w};
    *(uint2*)&out[(size_t)i * 4] = *(const uint2*)o;
}

// ---------------------------------------------------------------------------
// g[b,e] = (1/S) * sum_s sum_h hgate[(s*B+b),h] * gW2[e,h] + gb2[e]
// One block per b; 8 expert accumulators per thread.
// ---------------------------------------------------------------------------
__global__ __launch_bounds__(256)
void gate_reduce(const __bf16* __restrict__ hgate, const float* __restrict__ gW2,
                 const float* __restrict__ gb2, float* __restrict__ g)
{
    int b = blockIdx.x;
    int tid = threadIdx.x;
    float acc[NE] = {};
    for (int idx = tid; idx < NS * Hdim; idx += 256) {
        int srow = idx >> 10;
        int h    = idx & (Hdim - 1);
        float hv = (float)hgate[((size_t)(srow * NB + b) << 10) + h];
        #pragma unroll
        for (int e = 0; e < NE; ++e) acc[e] += hv * gW2[e * Hdim + h];
    }
    __shared__ float red[NE][256];
    #pragma unroll
    for (int e = 0; e < NE; ++e) red[e][tid] = acc[e];
    __syncthreads();
    for (int s2 = 128; s2 > 0; s2 >>= 1) {
        if (tid < s2) {
            #pragma unroll
            for (int e = 0; e < NE; ++e) red[e][tid] += red[e][tid + s2];
        }
        __syncthreads();
    }
    if (tid < NE) g[b * NE + tid] = red[tid][0] * (1.0f / (float)NS) + gb2[tid];
}

// ---------------------------------------------------------------------------
// Wm[b, :] = sum_e g[b,e] * eW[e, :]   (reads each expert element ONCE,
// produces all 8 batch outputs -> 268MB read total instead of 2.1GB)
// ---------------------------------------------------------------------------
__global__ __launch_bounds__(256)
void merge_weights(const float* __restrict__ eW, const float* __restrict__ g,
                   __bf16* __restrict__ Wm, int n_per4)
{
    __shared__ float gs[NB * NE];
    if (threadIdx.x < NB * NE) gs[threadIdx.x] = g[threadIdx.x];
    __syncthreads();

    int i = blockIdx.x * 256 + threadIdx.x;
    if (i >= n_per4) return;
    size_t idx   = (size_t)i * 4;
    size_t n_per = (size_t)n_per4 * 4;

    float4 ev[NE];
    #pragma unroll
    for (int e = 0; e < NE; ++e) ev[e] = *(const float4*)(eW + (size_t)e * n_per + idx);

    #pragma unroll
    for (int b = 0; b < NB; ++b) {
        float ax = 0.f, ay = 0.f, az = 0.f, aw = 0.f;
        #pragma unroll
        for (int e = 0; e < NE; ++e) {
            float gv = gs[b * NE + e];
            ax += gv * ev[e].x; ay += gv * ev[e].y;
            az += gv * ev[e].z; aw += gv * ev[e].w;
        }
        alignas(8) __bf16 o[4] = {(__bf16)ax, (__bf16)ay, (__bf16)az, (__bf16)aw};
        *(uint2*)(Wm + (size_t)b * n_per + idx) = *(const uint2*)o;
    }
}

__global__ __launch_bounds__(256)
void merge_bias(const float* __restrict__ eb, const float* __restrict__ g,
                float* __restrict__ bm, int n)
{
    int i = blockIdx.x * 256 + threadIdx.x;
    if (i >= n) return;
    #pragma unroll
    for (int b = 0; b < NB; ++b) {
        float a = 0.f;
        #pragma unroll
        for (int e = 0; e < NE; ++e) a += g[b * NE + e] * eb[(size_t)e * n + i];
        bm[(size_t)b * n + i] = a;
    }
}

// ---------------------------------------------------------------------------
extern "C" void kernel_launch(void* const* d_in, const int* in_sizes, int n_in,
                              void* d_out, int out_size, void* d_ws, size_t ws_size,
                              hipStream_t stream)
{
    const float* hs  = (const float*)d_in[0];   // [S,B,H] == rows r=s*B+b
    const float* gW1 = (const float*)d_in[1];   // [H,H]
    const float* gb1 = (const float*)d_in[2];   // [H]
    const float* gW2 = (const float*)d_in[3];   // [E,H]
    const float* gb2 = (const float*)d_in[4];   // [E]
    const float* eW1 = (const float*)d_in[5];   // [E,DFF,H]
    const float* eb1 = (const float*)d_in[6];   // [E,DFF]
    const float* eW2 = (const float*)d_in[7];   // [E,H,DFF]
    const float* eb2 = (const float*)d_in[8];   // [E,H]
    float* out = (float*)d_out;                 // [S,B,H]

    // ---- workspace carve-up (256B aligned) ----
    char*  ws  = (char*)d_ws;
    size_t off = 0;
    auto carve = [&](size_t bytes) -> void* {
        void* p = ws + off;
        off += (bytes + 255) & ~(size_t)255;
        return p;
    };
    const size_t M_gate = (size_t)NS * NB;             // 4096
    __bf16* xbf    = (__bf16*)carve(M_gate * Hdim * 2);        // 8.4 MB
    __bf16* gW1bf  = (__bf16*)carve((size_t)Hdim * Hdim * 2);  // 2.1 MB
    __bf16* hgate  = (__bf16*)carve(M_gate * Hdim * 2);        // 8.4 MB
    float*  g      = (float*) carve(NB * NE * 4);
    __bf16* W1m    = (__bf16*)carve((size_t)NB * DFF * Hdim * 2);  // 67 MB
    __bf16* W2m    = (__bf16*)carve((size_t)NB * Hdim * DFF * 2);  // 67 MB
    float*  b1m    = (float*) carve((size_t)NB * DFF * 4);
    float*  b2m    = (float*) carve((size_t)NB * Hdim * 4);
    __bf16* y1     = (__bf16*)carve((size_t)NB * NS * DFF * 2);    // 33.6 MB
    (void)ws_size; (void)in_sizes; (void)n_in; (void)out_size;

    // 1) converts
    {
        int n4 = (int)(M_gate * Hdim / 4);
        f32_to_bf16_vec4<<<(n4 + 255) / 256, 256, 0, stream>>>(hs, xbf, n4);
        int m4 = Hdim * Hdim / 4;
        f32_to_bf16_vec4<<<(m4 + 255) / 256, 256, 0, stream>>>(gW1, gW1bf, m4);
    }

    // 2) gate fc1: hgate = relu(x @ gW1^T + gb1), M=4096, N=1024, K=1024
    gemm_bf16_wmma<<<dim3(Hdim / TILE_N, (int)M_gate / TILE_M), 256, 0, stream>>>(
        xbf, Hdim, gW1bf, Hdim, gb1, hgate, Hdim, Hdim, /*relu_bf16*/1);

    // 3) gate reduce -> g[B,E]
    gate_reduce<<<NB, 256, 0, stream>>>(hgate, gW2, gb2, g);

    // 4) merged expert params
    {
        int n4 = DFF * Hdim / 4;  // 1,048,576 per expert / 4
        merge_weights<<<(n4 + 255) / 256, 256, 0, stream>>>(eW1, g, W1m, n4);
        merge_weights<<<(n4 + 255) / 256, 256, 0, stream>>>(eW2, g, W2m, n4);
        merge_bias<<<(DFF + 255) / 256, 256, 0, stream>>>(eb1, g, b1m, DFF);
        merge_bias<<<(Hdim + 255) / 256, 256, 0, stream>>>(eb2, g, b2m, Hdim);
    }

    // 5) merged MLP, per batch b: x_b rows are r = s*B+b (base b*H, lda B*H)
    for (int b = 0; b < NB; ++b) {
        // y1_b = relu(x_b @ W1m_b^T + b1m_b): M=512, N=4096, K=1024
        gemm_bf16_wmma<<<dim3(DFF / TILE_N, NS / TILE_M), 256, 0, stream>>>(
            xbf + (size_t)b * Hdim, NB * Hdim,
            W1m + (size_t)b * DFF * Hdim, Hdim,
            b1m + (size_t)b * DFF,
            y1 + (size_t)b * NS * DFF, DFF,
            Hdim, /*relu_bf16*/1);
    }
    for (int b = 0; b < NB; ++b) {
        // out_b = y1_b @ W2m_b^T + b2m_b (fp32): M=512, N=1024, K=4096
        gemm_bf16_wmma<<<dim3(Hdim / TILE_N, NS / TILE_M), 256, 0, stream>>>(
            y1 + (size_t)b * NS * DFF, DFF,
            W2m + (size_t)b * Hdim * DFF, DFF,
            b2m + (size_t)b * Hdim,
            out + (size_t)b * Hdim, NB * Hdim,
            DFF, /*fp32 out*/0);
    }
}
